// KimiDeltaAttention_62130996904031
// MI455X (gfx1250) — matrix operand order repaired and checked by reading, not compiled, not run on hardware
//
#include <hip/hip_runtime.h>

// ---------------------------------------------------------------------------
// Types for CDNA5 WMMA (wave32): v_wmma_f32_16x16x32_bf16
// ---------------------------------------------------------------------------
typedef __attribute__((ext_vector_type(8)))  __bf16 v8bf;
typedef __attribute__((ext_vector_type(16))) __bf16 v16bf;
typedef __attribute__((ext_vector_type(8)))  float  v8f;
typedef int v4i_vs __attribute__((vector_size(16)));   // matches builtin param

// Problem constants (from reference setup_inputs)
#define BB   2
#define TT   2048
#define DD   2048
#define HH   16
#define DK   128
#define DV   128
#define MT   (BB*TT)      // 4096 rows of the "token" dimension
#define HD   (HH*DK)      // 2048

// ---------------------------------------------------------------------------
// Async global->LDS staging (gfx1250 ASYNCcnt path) with compile-safe fallback
// ---------------------------------------------------------------------------
#if defined(__has_builtin)
#  if __has_builtin(__builtin_amdgcn_global_load_async_to_lds_b128) && \
      __has_builtin(__builtin_amdgcn_s_wait_asynccnt)
#    define KDA_ASYNC 1
#  endif
#endif
#ifndef KDA_ASYNC
#  define KDA_ASYNC 0
#endif

__device__ __forceinline__ void stage16(const __bf16* g, __bf16* l) {
#if KDA_ASYNC
  // global_load_async_to_lds_b128 (tracked by ASYNCcnt)
  __builtin_amdgcn_global_load_async_to_lds_b128(
      (__attribute__((address_space(1))) v4i_vs*)(g),
      (__attribute__((address_space(3))) v4i_vs*)(l),
      0, 0);
#else
  *(v8bf*)l = *(const v8bf*)g;    // load + ds_store fallback
#endif
}

template <int N>
__device__ __forceinline__ void await_stage() {
#if KDA_ASYNC
  __builtin_amdgcn_s_wait_asynccnt(N);
#endif
}

// ---------------------------------------------------------------------------
// Elementwise f32 -> bf16 conversion
// ---------------------------------------------------------------------------
__global__ void __launch_bounds__(256) cvt_bf16_kernel(
    const float* __restrict__ src, __bf16* __restrict__ dst, size_t n) {
  size_t i = (size_t)blockIdx.x * 256 + threadIdx.x;
  if (i < n) dst[i] = (__bf16)src[i];
}

// W [K,N] f32 row-major  ->  WT [N,K] bf16 row-major (so B fragments are
// contiguous K-runs per lane, matching the ISA 16-bit B-matrix layout).
__global__ void __launch_bounds__(256) transpose_cvt_bf16_kernel(
    const float* __restrict__ W, __bf16* __restrict__ WT, int K, int N) {
  size_t i = (size_t)blockIdx.x * 256 + threadIdx.x;
  if (i >= (size_t)K * N) return;
  int k = (int)(i % K);
  int n = (int)(i / K);
  WT[i] = (__bf16)W[(size_t)k * N + n];
}

// ---------------------------------------------------------------------------
// bf16 WMMA GEMM:  C[M,N] f32 = A[M,K]bf16 (row-major) * BT[N,K]bf16 + bias
//
// Block = 256 threads (8 waves), block tile 128x128, K-step 32.
// A/B tiles double-buffered in LDS (2 x (8KB + 8KB) = 32KB), staged with
// GLOBAL_LOAD_ASYNC_TO_LDS_B128 (ASYNCcnt) when available.
// Wave grid 4x2; each wave computes 32x64 = 2x4 tiles of 16x16x32 WMMA.
//
// A fragment per lane (ISA 7.12.2, 16-bit A 16x32): row = lane%16,
//   K = (lane/16)*8 .. +7  and  16+(lane/16)*8 .. +7.
// B fragment per lane (16-bit B 32x16): col = lane%16,
//   K = (lane/16)*8 .. +7  and  16+(lane/16)*8 .. +7 (same as A).
// ---------------------------------------------------------------------------
__device__ __forceinline__ v16bf cat8(v8bf lo, v8bf hi) {
  return __builtin_shufflevector(lo, hi, 0,1,2,3,4,5,6,7,8,9,10,11,12,13,14,15);
}

__global__ void __launch_bounds__(256) gemm_bf16_wmma(
    const __bf16* __restrict__ A, const __bf16* __restrict__ BT,
    const float* __restrict__ bias, float* __restrict__ C,
    int M, int N, int K) {
  __shared__ __bf16 lA[2][128][32];
  __shared__ __bf16 lB[2][128][32];

  const int tid  = threadIdx.x;
  const int lane = tid & 31;
  const int wave = tid >> 5;
  const int wm   = wave & 3;        // 0..3 -> rows
  const int wn   = wave >> 2;       // 0..1 -> cols
  const int blockM = blockIdx.x * 128;
  const int blockN = blockIdx.y * 128;
  const int l15 = lane & 15;
  const int lhi = lane >> 4;

  const v8f vzero = {0.f,0.f,0.f,0.f,0.f,0.f,0.f,0.f};
  v8f acc[2][4];
#pragma unroll
  for (int m = 0; m < 2; ++m)
#pragma unroll
    for (int n = 0; n < 4; ++n) acc[m][n] = vzero;

  // Staging: per tile, 128 rows x 4 chunks of 16B; 256 threads -> 2 chunks
  // each per array (4 async b128 issues per thread per K-step).
  auto stage_tile = [&](int buf, int k0) {
#pragma unroll
    for (int rep = 0; rep < 2; ++rep) {
      int id  = rep * 256 + tid;
      int row = id >> 2;
      int kc  = (id & 3) * 8;
      int ra  = blockM + row; if (ra >= M) ra = M - 1;
      stage16(A + (size_t)ra * K + k0 + kc, &lA[buf][row][kc]);
      int rb  = blockN + row; if (rb >= N) rb = N - 1;
      stage16(BT + (size_t)rb * K + k0 + kc, &lB[buf][row][kc]);
    }
  };

  const int nk = K >> 5;            // K / 32
  stage_tile(0, 0);

  for (int kt = 0; kt < nk; ++kt) {
    const int cur = kt & 1;
    __syncthreads();                       // buf[1-cur] readers (kt-1) done
    if (kt + 1 < nk) {
      stage_tile(1 - cur, (kt + 1) * 32);  // overlap DMA with WMMA
      if (kt + 2 < nk) {                   // prefetch tile kt+2 into L2
        __builtin_prefetch(A + (size_t)(blockM + (tid >> 1)) * K + (kt + 2) * 32, 0, 1);
      }
      await_stage<4>();                    // 4 newest (next tile) may remain
    } else {
      await_stage<0>();
    }
    __syncthreads();                       // staged data visible to all waves

    v16bf af[2], bfrag[4];
#pragma unroll
    for (int m = 0; m < 2; ++m) {
      const __bf16* p = &lA[cur][wm * 32 + m * 16 + l15][lhi * 8];
      af[m] = cat8(*(const v8bf*)p, *(const v8bf*)(p + 16));
    }
#pragma unroll
    for (int n = 0; n < 4; ++n) {
      const __bf16* p = &lB[cur][wn * 64 + n * 16 + l15][lhi * 8];
      bfrag[n] = cat8(*(const v8bf*)p, *(const v8bf*)(p + 16));
    }
#pragma unroll
    for (int m = 0; m < 2; ++m)
#pragma unroll
      for (int n = 0; n < 4; ++n)
        acc[m][n] = __builtin_amdgcn_wmma_f32_16x16x32_bf16(
            false, af[m], false, bfrag[n], (short)0, acc[m][n], false, false);
  }

  // C/D 16x16 f32 layout: lane<16 -> rows 0..7 (V0..7), lane>=16 -> rows 8..15
  const int rowBase = blockM + wm * 32;
  const int colBase = blockN + wn * 64;
  const int srow = lhi * 8;
  const int scol = l15;
#pragma unroll
  for (int m = 0; m < 2; ++m) {
#pragma unroll
    for (int n = 0; n < 4; ++n) {
      int c = colBase + n * 16 + scol;
      if (c >= N) continue;
      float bv = bias ? bias[c] : 0.f;
#pragma unroll
      for (int v = 0; v < 8; ++v) {
        int r = rowBase + m * 16 + srow + v;
        if (r < M) C[(size_t)r * N + c] = acc[m][n][v] + bv;
      }
    }
  }
}

// ---------------------------------------------------------------------------
// Causal depthwise conv (K=4) + SiLU.  X: [B,T,C], W4: [C,4]
// ---------------------------------------------------------------------------
__global__ void __launch_bounds__(256) conv_silu_kernel(
    const float* __restrict__ X, const float* __restrict__ W4,
    float* __restrict__ Y) {
  size_t i = (size_t)blockIdx.x * 256 + threadIdx.x;
  if (i >= (size_t)BB * TT * DD) return;
  int c = (int)(i % DD);
  size_t bt = i / DD;
  int t = (int)(bt % TT);
  size_t b = bt / TT;
  float acc = 0.f;
#pragma unroll
  for (int kk = 0; kk < 4; ++kk) {
    int tt = t - 3 + kk;
    if (tt >= 0) acc = fmaf(X[(b * TT + tt) * (size_t)DD + c], W4[c * 4 + kk], acc);
  }
  Y[i] = acc / (1.f + __expf(-acc));   // SiLU
}

// ---------------------------------------------------------------------------
// prep: per (b,t,h) block of 128 threads:
//   q,k  <- l2norm (in-place);  g2 <- exp(-exp(A_log)*softplus(g2+dt_bias));
//   beta <- sigmoid(bl)
// ---------------------------------------------------------------------------
__global__ void __launch_bounds__(128) prep_kernel(
    float* __restrict__ qc, float* __restrict__ kc, float* __restrict__ g2,
    const float* __restrict__ bl, float* __restrict__ beta,
    const float* __restrict__ A_log, const float* __restrict__ dtb) {
  __shared__ float red[128];
  const int idx = blockIdx.x;           // bt*H + h
  const int h   = idx & (HH - 1);
  const int c   = threadIdx.x;
  const size_t base = (size_t)idx * DK + c;

  float qv = qc[base];
  red[c] = qv * qv; __syncthreads();
  for (int s = 64; s > 0; s >>= 1) { if (c < s) red[c] += red[c + s]; __syncthreads(); }
  float qn = qv * rsqrtf(red[0] + 1e-6f);
  __syncthreads();

  float kv = kc[base];
  red[c] = kv * kv; __syncthreads();
  for (int s = 64; s > 0; s >>= 1) { if (c < s) red[c] += red[c + s]; __syncthreads(); }
  float kn = kv * rsqrtf(red[0] + 1e-6f);

  qc[base] = qn;
  kc[base] = kn;

  float gv = g2[base] + dtb[h * DK + c];
  float sp = (gv > 20.f) ? gv : log1pf(__expf(gv));
  g2[base] = __expf(-__expf(A_log[h]) * sp);

  if (c == 0) beta[idx] = 1.f / (1.f + __expf(-bl[idx]));
}

// ---------------------------------------------------------------------------
// Sequential delta-rule scan.  One workgroup per (b,h).  State S[Dk][Dv]
// lives in registers: thread (j, yi) owns S[yi*64 .. yi*64+63][j] (64 f32).
// Cross-thread kT*S and qT*S reductions go through LDS.
// ---------------------------------------------------------------------------
__global__ void __launch_bounds__(256) kda_scan_kernel(
    const float* __restrict__ q, const float* __restrict__ k,
    const float* __restrict__ v, const float* __restrict__ ge,
    const float* __restrict__ beta, float* __restrict__ o) {
  const int bh = blockIdx.x;            // 0..B*H-1
  const int b  = bh >> 4;
  const int h  = bh & (HH - 1);
  const int j  = threadIdx.x;           // 0..127  (Dv column)
  const int yi = threadIdx.y;           // 0..1    (Dk half)
  const int tid = yi * 128 + j;

  __shared__ float sq[DK], sk[DK], sg[DK], sv[DV];
  __shared__ float red[2][DV];
  __shared__ float sbeta;

  float S[64];
#pragma unroll
  for (int ii = 0; ii < 64; ++ii) S[ii] = 0.f;
  const float scale = 0.08838834764831845f;   // Dk^-0.5

  for (int t = 0; t < TT; ++t) {
    const size_t base = ((size_t)(b * TT + t) * HH + h) * DK;
    if (yi == 0) { sq[j] = q[base + j]; sv[j] = v[base + j]; }
    else         { sk[j] = k[base + j]; sg[j] = ge[base + j]; }
    if (tid == 0) sbeta = beta[(size_t)(b * TT + t) * HH + h];
    __syncthreads();

    // decay + partial kT*S
    float kvp = 0.f;
#pragma unroll
    for (int ii = 0; ii < 64; ++ii) {
      int i = yi * 64 + ii;
      float s = S[ii] * sg[i];
      S[ii] = s;
      kvp = fmaf(sk[i], s, kvp);
    }
    red[yi][j] = kvp;
    __syncthreads();

    float upd = (sv[j] - (red[0][j] + red[1][j])) * sbeta;

    // rank-1 update + partial qT*S
    float op = 0.f;
#pragma unroll
    for (int ii = 0; ii < 64; ++ii) {
      int i = yi * 64 + ii;
      float s = fmaf(sk[i], upd, S[ii]);
      S[ii] = s;
      op = fmaf(sq[i], s, op);
    }
    __syncthreads();          // all reads of red / shared done
    red[yi][j] = op;
    __syncthreads();
    if (yi == 0) o[base + j] = (red[0][j] + red[1][j]) * scale;
    __syncthreads();          // before next iteration's shared writes
  }
}

// ---------------------------------------------------------------------------
// Gated RMSNorm over Dv per (b,t,h), * silu(gate), emit bf16 for final GEMM.
// ---------------------------------------------------------------------------
__global__ void __launch_bounds__(128) norm_gate_kernel(
    const float* __restrict__ o, const float* __restrict__ gl,
    const float* __restrict__ onw, __bf16* __restrict__ onb) {
  __shared__ float red[128];
  const int idx = blockIdx.x;
  const int c   = threadIdx.x;
  const size_t base = (size_t)idx * DV + c;
  float ov = o[base];
  red[c] = ov * ov; __syncthreads();
  for (int s = 64; s > 0; s >>= 1) { if (c < s) red[c] += red[c + s]; __syncthreads(); }
  float on = ov * rsqrtf(red[0] * (1.f / DV) + 1e-5f) * onw[c];
  float gp = gl[base];
  on *= gp / (1.f + __expf(-gp));     // * silu(gate)
  onb[base] = (__bf16)on;
}

// ---------------------------------------------------------------------------
// Host-side orchestration
// ---------------------------------------------------------------------------
static inline dim3 grid1(size_t n) { return dim3((unsigned)((n + 255) / 256)); }

extern "C" void kernel_launch(void* const* d_in, const int* in_sizes, int n_in,
                              void* d_out, int out_size, void* d_ws, size_t ws_size,
                              hipStream_t stream) {
  const float* x     = (const float*)d_in[0];
  const float* Wq    = (const float*)d_in[1];
  const float* Wk    = (const float*)d_in[2];
  const float* Wv    = (const float*)d_in[3];
  const float* cq    = (const float*)d_in[4];
  const float* ck    = (const float*)d_in[5];
  const float* cv    = (const float*)d_in[6];
  const float* Wf1   = (const float*)d_in[7];
  const float* Wf2   = (const float*)d_in[8];
  const float* Wb    = (const float*)d_in[9];
  const float* A_log = (const float*)d_in[10];
  const float* dtb   = (const float*)d_in[11];
  const float* Wg    = (const float*)d_in[12];
  const float* bg    = (const float*)d_in[13];
  const float* onw   = (const float*)d_in[14];
  const float* Wo    = (const float*)d_in[15];
  float* out = (float*)d_out;

  char* wsb = (char*)d_ws;
  size_t off = 0;
  auto alloc = [&](size_t bytes) -> void* {
    void* p = wsb + off;
    off += (bytes + 255) & ~(size_t)255;
    return p;
  };

  __bf16* XB   = (__bf16*)alloc((size_t)MT * DD * 2);     // bf16 activations
  __bf16* WqT  = (__bf16*)alloc((size_t)HD * DD * 2);
  __bf16* WkT  = (__bf16*)alloc((size_t)HD * DD * 2);
  __bf16* WvT  = (__bf16*)alloc((size_t)HD * DD * 2);
  __bf16* WgT  = (__bf16*)alloc((size_t)HD * DD * 2);
  __bf16* WoT  = (__bf16*)alloc((size_t)DD * HD * 2);
  __bf16* Wf1T = (__bf16*)alloc((size_t)DV * DD * 2);
  __bf16* Wf2T = (__bf16*)alloc((size_t)HD * DV * 2);
  __bf16* WbT  = (__bf16*)alloc((size_t)HH * DD * 2);
  float*  QL   = (float*) alloc((size_t)MT * HD * 4);
  float*  KL   = (float*) alloc((size_t)MT * HD * 4);
  float*  VL   = (float*) alloc((size_t)MT * HD * 4);
  float*  GL   = (float*) alloc((size_t)MT * HD * 4);
  float*  BL   = (float*) alloc((size_t)MT * HH * 4);
  float*  F1   = (float*) alloc((size_t)MT * DV * 4);
  __bf16* F1B  = (__bf16*)alloc((size_t)MT * DV * 2);
  float*  G2   = (float*) alloc((size_t)MT * HD * 4);
  float*  QC   = (float*) alloc((size_t)MT * HD * 4);
  float*  BETA = (float*) alloc((size_t)MT * HH * 4);
  // aliases (dead buffers reused)
  float*  KC  = QL;              // conv_k output (QL dead after conv_q)
  float*  VC  = KL;              // conv_v output (KL dead after conv_k)
  float*  O   = VL;              // scan output (VL dead after conv_v)
  __bf16* ONB = XB;              // gated-norm bf16 output (XB dead by then)

  // 1) activations -> bf16
  cvt_bf16_kernel<<<grid1((size_t)MT * DD), 256, 0, stream>>>(x, XB, (size_t)MT * DD);

  // 2) weights -> transposed bf16
  transpose_cvt_bf16_kernel<<<grid1((size_t)DD * HD), 256, 0, stream>>>(Wq,  WqT,  DD, HD);
  transpose_cvt_bf16_kernel<<<grid1((size_t)DD * HD), 256, 0, stream>>>(Wk,  WkT,  DD, HD);
  transpose_cvt_bf16_kernel<<<grid1((size_t)DD * HD), 256, 0, stream>>>(Wv,  WvT,  DD, HD);
  transpose_cvt_bf16_kernel<<<grid1((size_t)DD * HD), 256, 0, stream>>>(Wg,  WgT,  DD, HD);
  transpose_cvt_bf16_kernel<<<grid1((size_t)HD * DD), 256, 0, stream>>>(Wo,  WoT,  HD, DD);
  transpose_cvt_bf16_kernel<<<grid1((size_t)DD * DV), 256, 0, stream>>>(Wf1, Wf1T, DD, DV);
  transpose_cvt_bf16_kernel<<<grid1((size_t)DV * HD), 256, 0, stream>>>(Wf2, Wf2T, DV, HD);
  transpose_cvt_bf16_kernel<<<grid1((size_t)DD * HH), 256, 0, stream>>>(Wb,  WbT,  DD, HH);

  // 3) projection GEMMs (WMMA bf16, async-LDS double buffered)
  dim3 blk(256);
  dim3 gBig(MT / 128, HD / 128);
  gemm_bf16_wmma<<<gBig, blk, 0, stream>>>(XB, WqT, nullptr, QL, MT, HD, DD);
  gemm_bf16_wmma<<<gBig, blk, 0, stream>>>(XB, WkT, nullptr, KL, MT, HD, DD);
  gemm_bf16_wmma<<<gBig, blk, 0, stream>>>(XB, WvT, nullptr, VL, MT, HD, DD);
  gemm_bf16_wmma<<<gBig, blk, 0, stream>>>(XB, WgT, bg,      GL, MT, HD, DD);
  gemm_bf16_wmma<<<dim3(MT / 128, 1), blk, 0, stream>>>(XB, WbT,  nullptr, BL, MT, HH, DD);
  gemm_bf16_wmma<<<dim3(MT / 128, 1), blk, 0, stream>>>(XB, Wf1T, nullptr, F1, MT, DV, DD);

  // 4) g2 = (f1 bf16) @ Wf2
  cvt_bf16_kernel<<<grid1((size_t)MT * DV), 256, 0, stream>>>(F1, F1B, (size_t)MT * DV);
  gemm_bf16_wmma<<<gBig, blk, 0, stream>>>(F1B, Wf2T, nullptr, G2, MT, HD, DV);

  // 5) causal depthwise conv + SiLU (order matters for buffer aliasing)
  conv_silu_kernel<<<grid1((size_t)MT * DD), 256, 0, stream>>>(QL, cq, QC);
  conv_silu_kernel<<<grid1((size_t)MT * DD), 256, 0, stream>>>(KL, ck, KC);
  conv_silu_kernel<<<grid1((size_t)MT * DD), 256, 0, stream>>>(VL, cv, VC);

  // 6) l2norm q/k, decay gexp, beta
  prep_kernel<<<MT * HH, 128, 0, stream>>>(QC, KC, G2, BL, BETA, A_log, dtb);

  // 7) sequential delta-rule scan (one WGP workgroup per (b,h))
  kda_scan_kernel<<<BB * HH, dim3(128, 2), 0, stream>>>(QC, KC, VC, G2, BETA, O);

  // 8) gated RMSNorm -> bf16
  norm_gate_kernel<<<MT * HH, 128, 0, stream>>>(O, GL, onw, ONB);

  // 9) final projection: out = on @ Wo
  gemm_bf16_wmma<<<dim3(MT / 128, DD / 128), blk, 0, stream>>>(ONB, WoT, nullptr, out, MT, DD, HD);
}
